// AttentionBlock_7456063226402
// MI455X (gfx1250) — matrix operand order repaired and checked by reading, not compiled, not run on hardware
//
#include <hip/hip_runtime.h>

typedef __attribute__((ext_vector_type(16))) _Float16 v16h;
typedef __attribute__((ext_vector_type(8)))  float    v8f;

#define B_    8
#define C_    512
#define T_    1024
#define G_    32
#define CPG_  16
#define NH_   8
#define CH_   64
#define BH_   64      // B_*NH_
#define OC3_  1536    // 3*C_
#define QK_SCALE 0.35355339059327373f   // 1/sqrt(sqrt(64))
#define EPS_  1e-5f

#define NWQ_  (OC3_ * C_)   // 786432 qkv_w elements
#define NWP_  (C_ * C_)     // 262144 proj_w elements

// A-matrix (f16, 16x32 MxK) element->K mapping (matches B fill order).
// e=0..15 -> K = half*16 + e           (contiguous 16-half run)
__device__ __forceinline__ int kA(int e, int half) {
    return half * 16 + e;
}

__device__ __forceinline__ v8f wmma_f16(v16h a, v16h b, v8f c) {
    return __builtin_amdgcn_wmma_f32_16x16x32_f16(
        /*neg_a=*/false, a, /*neg_b=*/false, b,
        /*c_mod=*/(short)0, c, /*reuse_a=*/false, /*reuse_b=*/false);
}

// ---------------------------------------------------------------------------
// Weight pre-convert f32 -> f16 (one pass; removes cvts from GEMM hot loops).
// ---------------------------------------------------------------------------
__global__ __launch_bounds__(256)
void wcvt_kernel(const float* __restrict__ wq,
                 const float* __restrict__ wp,
                 _Float16* __restrict__ wq16,
                 _Float16* __restrict__ wp16) {
    const int i = blockIdx.x * 256 + threadIdx.x;
    if (i < NWQ_) wq16[i] = (_Float16)wq[i];
    if (i < NWP_) wp16[i] = (_Float16)wp[i];
}

// ---------------------------------------------------------------------------
// GroupNorm: one block per (batch, group); writes T-major f16 xnT[b][t][c].
// ---------------------------------------------------------------------------
__global__ __launch_bounds__(256)
void gn_kernel(const float* __restrict__ x,
               const float* __restrict__ gamma,
               const float* __restrict__ beta,
               _Float16* __restrict__ xnT) {
    const int bg   = blockIdx.x;            // [0, B_*G_)
    const int b    = bg / G_, g = bg % G_;
    const int base = b * C_ * T_ + g * CPG_ * T_;
    const int N    = CPG_ * T_;             // 16384
    const int tid  = threadIdx.x;

    // coalesced reduction pass (channel-major indexing)
    float s = 0.f, s2 = 0.f;
    for (int i = tid; i < N; i += 256) {
        float v = x[base + i];
        s += v; s2 += v * v;
    }
    for (int msk = 16; msk >= 1; msk >>= 1) {
        s  += __shfl_xor(s,  msk, 32);
        s2 += __shfl_xor(s2, msk, 32);
    }
    __shared__ float ws1[8], ws2[8], stat[2];
    const int wave = tid >> 5, lane = tid & 31;
    if (lane == 0) { ws1[wave] = s; ws2[wave] = s2; }
    __syncthreads();
    if (tid == 0) {
        float a = 0.f, a2 = 0.f;
        for (int i = 0; i < 8; ++i) { a += ws1[i]; a2 += ws2[i]; }
        float mean = a / (float)N;
        float var  = a2 / (float)N - mean * mean;
        stat[0] = mean;
        stat[1] = rsqrtf(var + EPS_);
    }
    __syncthreads();
    const float mean = stat[0], rstd = stat[1];
    // normalize pass indexed t-major so xnT writes are contiguous in c
    for (int i = tid; i < N; i += 256) {
        const int t  = i >> 4;
        const int cc = i & 15;
        const int c  = g * CPG_ + cc;
        float v = (x[base + cc * T_ + t] - mean) * rstd * gamma[c] + beta[c];
        xnT[((size_t)b * T_ + t) * C_ + c] = (_Float16)v;
    }
}

// ---------------------------------------------------------------------------
// QKV GEMM: qkv[b,o,t] = sum_c W[o,c]*xn[c,t] + bias[o].
// One wave per 16(o) x 16(t) tile. Each 16-wide o-tile lies entirely in one
// q/k/v third (192 = 12*16) -> block-uniform routing.
//   q,k -> T-major [bh][t][ch] (pre-scaled), v -> [bh][ch][t].
// ---------------------------------------------------------------------------
__global__ __launch_bounds__(32)
void qkv_kernel(const _Float16* __restrict__ xnT,
                const _Float16* __restrict__ w16,
                const float* __restrict__ bias,
                _Float16* __restrict__ qhT,
                _Float16* __restrict__ khT,
                _Float16* __restrict__ vh) {
    const int o0   = blockIdx.x * 16;
    const int t0   = blockIdx.y * 16;
    const int b    = blockIdx.z;
    const int lane = threadIdx.x & 31;
    const int half = lane >> 4;
    const int mn   = lane & 15;

    v8f acc = {};
    for (int kc = 0; kc < C_ / 32; ++kc) {
        const int c0 = kc * 32;
        v16h a, bb;
        const _Float16* ap = &w16[(size_t)(o0 + mn) * C_ + c0];
        const _Float16* bp = &xnT[((size_t)b * T_ + t0 + mn) * C_ + c0 + half * 16];
        #pragma unroll
        for (int e = 0; e < 16; ++e) a[e]  = ap[kA(e, half)];   // 2x16B runs
        #pragma unroll
        for (int e = 0; e < 16; ++e) bb[e] = bp[e];             // 32B run
        acc = wmma_f16(a, bb, acc);
    }

    const int head = o0 / 192, r0 = o0 % 192;
    const int bh   = b * NH_ + head;
    const int t    = t0 + mn;
    if (r0 < 64) {                      // q (scaled), T-major
        _Float16* dst = &qhT[((size_t)bh * T_ + t) * CH_ + r0 + 8 * half];
        #pragma unroll
        for (int i = 0; i < 8; ++i)
            dst[i] = (_Float16)((acc[i] + bias[o0 + i + 8 * half]) * QK_SCALE);
    } else if (r0 < 128) {              // k (scaled), T-major
        _Float16* dst = &khT[((size_t)bh * T_ + t) * CH_ + (r0 - 64) + 8 * half];
        #pragma unroll
        for (int i = 0; i < 8; ++i)
            dst[i] = (_Float16)((acc[i] + bias[o0 + i + 8 * half]) * QK_SCALE);
    } else {                            // v, channel-major
        #pragma unroll
        for (int i = 0; i < 8; ++i) {
            const int rv = (r0 - 128) + i + 8 * half;
            vh[((size_t)bh * CH_ + rv) * T_ + t] =
                (_Float16)(acc[i] + bias[o0 + i + 8 * half]);
        }
    }
}

// ---------------------------------------------------------------------------
// Attention: one block (4 waves) per (bh, 16-row t-tile).
// Phase A: scores (QK^T) -> LDS f16 (16x1024 = 32 KB).
// Phase B: softmax in place (f32 math, f16 storage).
// Phase C: h = V @ P^T  (M=ch, N=t) so V is the contiguous A operand and
//          probabilities load as 2x ds_load_b128; output stored T-major.
// ---------------------------------------------------------------------------
__global__ __launch_bounds__(128)
void attn_kernel(const _Float16* __restrict__ qhT,
                 const _Float16* __restrict__ khT,
                 const _Float16* __restrict__ vh,
                 _Float16* __restrict__ hhT) {
    __shared__ _Float16 sc[16 * T_];    // 32 KB of the 320 KB WGP LDS
    const int t0   = blockIdx.x * 16;
    const int bh   = blockIdx.y;
    const int b    = bh / NH_, head = bh % NH_;
    const int tid  = threadIdx.x;
    const int wave = tid >> 5;
    const int lane = tid & 31;
    const int half = lane >> 4;
    const int mn   = lane & 15;

    // Preload A = q^T tile (16 t x 64 ch) as two 16x32 f16 operands.
    v16h aq0, aq1;
    {
        const _Float16* qp = &qhT[((size_t)bh * T_ + t0 + mn) * CH_];
        #pragma unroll
        for (int e = 0; e < 16; ++e) {
            aq0[e] = qp[kA(e, half)];
            aq1[e] = qp[32 + kA(e, half)];
        }
    }

    // Phase A: each wave fills a 16 x 256 strip of scores.
    for (int j = 0; j < 16; ++j) {
        const int s0 = wave * 256 + j * 16;
        const _Float16* kp = &khT[((size_t)bh * T_ + s0 + mn) * CH_];
        v16h bk;
        v8f acc = {};
        #pragma unroll
        for (int e = 0; e < 16; ++e) bk[e] = kp[half * 16 + e];
        acc = wmma_f16(aq0, bk, acc);
        #pragma unroll
        for (int e = 0; e < 16; ++e) bk[e] = kp[32 + half * 16 + e];
        acc = wmma_f16(aq1, bk, acc);
        #pragma unroll
        for (int i = 0; i < 8; ++i)
            sc[(i + 8 * half) * T_ + s0 + mn] = (_Float16)acc[i];
    }
    __syncthreads();

    // Phase B: softmax per row, normalized in place (4 rows per wave).
    for (int j = 0; j < 4; ++j) {
        const int r = wave * 4 + j;
        float mx = -1e30f;
        for (int i = lane; i < T_; i += 32)
            mx = fmaxf(mx, (float)sc[r * T_ + i]);
        for (int msk = 16; msk >= 1; msk >>= 1)
            mx = fmaxf(mx, __shfl_xor(mx, msk, 32));
        float sum = 0.f;
        for (int i = lane; i < T_; i += 32) {
            float p = __expf((float)sc[r * T_ + i] - mx);
            sc[r * T_ + i] = (_Float16)p;
            sum += p;
        }
        for (int msk = 16; msk >= 1; msk >>= 1)
            sum += __shfl_xor(sum, msk, 32);
        const float inv = 1.0f / sum;
        for (int i = lane; i < T_; i += 32)
            sc[r * T_ + i] = (_Float16)((float)sc[r * T_ + i] * inv);
    }
    __syncthreads();

    // Phase C: D[ch, t] = sum_s V[ch,s] * P[t,s]; wave owns 16-ch tile.
    const int ch0 = wave * 16;
    v8f acc = {};
    for (int ks = 0; ks < T_ / 32; ++ks) {
        const int s0 = ks * 32;
        v16h a, bb;
        const _Float16* vp = &vh[((size_t)bh * CH_ + ch0 + mn) * T_ + s0];
        const _Float16* pp = &sc[mn * T_ + s0 + half * 16];
        #pragma unroll
        for (int e = 0; e < 16; ++e) a[e]  = vp[kA(e, half)];   // 2x16B runs
        #pragma unroll
        for (int e = 0; e < 16; ++e) bb[e] = pp[e];             // 2x ds b128
        acc = wmma_f16(a, bb, acc);
    }
    // D element (m=ch_local, n=t_local): store T-major, 8 consecutive halves.
    {
        _Float16* dst = &hhT[((size_t)b * T_ + t0 + mn) * C_ +
                             head * CH_ + ch0 + 8 * half];
        #pragma unroll
        for (int i = 0; i < 8; ++i) dst[i] = (_Float16)acc[i];
    }
}

// ---------------------------------------------------------------------------
// Projection + residual: out[b,o,t] = x[b,o,t] + sum_c P[o,c]*h[b,c,t] + pb[o]
// ---------------------------------------------------------------------------
__global__ __launch_bounds__(32)
void proj_kernel(const _Float16* __restrict__ hhT,
                 const _Float16* __restrict__ w16,
                 const float* __restrict__ bias,
                 const float* __restrict__ x,
                 float* __restrict__ out) {
    const int o0   = blockIdx.x * 16;
    const int t0   = blockIdx.y * 16;
    const int b    = blockIdx.z;
    const int lane = threadIdx.x & 31;
    const int half = lane >> 4;
    const int mn   = lane & 15;

    v8f acc = {};
    for (int kc = 0; kc < C_ / 32; ++kc) {
        const int c0 = kc * 32;
        v16h a, bb;
        const _Float16* ap = &w16[(size_t)(o0 + mn) * C_ + c0];
        const _Float16* bp = &hhT[((size_t)b * T_ + t0 + mn) * C_ + c0 + half * 16];
        #pragma unroll
        for (int e = 0; e < 16; ++e) a[e]  = ap[kA(e, half)];
        #pragma unroll
        for (int e = 0; e < 16; ++e) bb[e] = bp[e];
        acc = wmma_f16(a, bb, acc);
    }
    #pragma unroll
    for (int i = 0; i < 8; ++i) {
        const int o = o0 + i + 8 * half;
        const int t = t0 + mn;
        const size_t idx = ((size_t)b * C_ + o) * T_ + t;
        out[idx] = x[idx] + acc[i] + bias[o];
    }
}

extern "C" void kernel_launch(void* const* d_in, const int* in_sizes, int n_in,
                              void* d_out, int out_size, void* d_ws, size_t ws_size,
                              hipStream_t stream) {
    (void)in_sizes; (void)n_in; (void)out_size; (void)ws_size;
    const float* x    = (const float*)d_in[0];
    const float* gns  = (const float*)d_in[1];
    const float* gnb  = (const float*)d_in[2];
    const float* qkvw = (const float*)d_in[3];
    const float* qkvb = (const float*)d_in[4];
    const float* pw   = (const float*)d_in[5];
    const float* pb   = (const float*)d_in[6];
    float* out = (float*)d_out;

    // Workspace layout (f16 elements):
    //   xnT, qhT, khT, vh, hhT : B*C*T each (8 MB each)
    //   wq16 : 1536x512, wp16 : 512x512
    const size_t NE = (size_t)B_ * C_ * T_;
    _Float16* ws   = (_Float16*)d_ws;
    _Float16* xnT  = ws;
    _Float16* qhT  = ws + 1 * NE;
    _Float16* khT  = ws + 2 * NE;
    _Float16* vh   = ws + 3 * NE;
    _Float16* hhT  = ws + 4 * NE;
    _Float16* wq16 = ws + 5 * NE;
    _Float16* wp16 = wq16 + (size_t)NWQ_;

    hipLaunchKernelGGL(wcvt_kernel, dim3((NWQ_ + 255) / 256), dim3(256), 0, stream,
                       qkvw, pw, wq16, wp16);
    hipLaunchKernelGGL(gn_kernel, dim3(B_ * G_), dim3(256), 0, stream,
                       x, gns, gnb, xnT);
    hipLaunchKernelGGL(qkv_kernel, dim3(OC3_ / 16, T_ / 16, B_), dim3(32), 0, stream,
                       xnT, wq16, qkvb, qhT, khT, vh);
    hipLaunchKernelGGL(attn_kernel, dim3(T_ / 16, BH_), dim3(128), 0, stream,
                       qhT, khT, vh, hhT);
    hipLaunchKernelGGL(proj_kernel, dim3(C_ / 16, T_ / 16, B_), dim3(32), 0, stream,
                       hhT, wp16, pb, x, out);
}
